// CausalMultiHeadSelfAttention_29051158790358
// MI455X (gfx1250) — matrix operand order repaired and checked by reading, not compiled, not run on hardware
//
#include <hip/hip_runtime.h>

typedef __attribute__((ext_vector_type(16))) __bf16 v16bf;
typedef __attribute__((ext_vector_type(8)))  __bf16 v8bf;
typedef __attribute__((ext_vector_type(8)))  float  v8f;

#define NB 4
#define NS 2048
#define ND 1024
#define NH 16
#define NDK 64
#define NM (NB*NS)   // 8192 rows

static __device__ __forceinline__ v8f wmma_bf16(v16bf a, v16bf b, v8f c) {
  // D = A(16x32) * B(32x16) + C, f32 accumulate
  return __builtin_amdgcn_wmma_f32_16x16x32_bf16(false, a, false, b, (short)0, c, false, false);
}

// ---------------- fp32 -> bf16 convert (vectorized x4) ----------------
__global__ void cvt_f32_to_bf16(const float4* __restrict__ src,
                                ushort4* __restrict__ dst, int n4) {
  int i = blockIdx.x * blockDim.x + threadIdx.x;
  if (i >= n4) return;
  float4 v = src[i];
  ushort4 o;
  o.x = __builtin_bit_cast(unsigned short, (__bf16)v.x);
  o.y = __builtin_bit_cast(unsigned short, (__bf16)v.y);
  o.z = __builtin_bit_cast(unsigned short, (__bf16)v.z);
  o.w = __builtin_bit_cast(unsigned short, (__bf16)v.w);
  dst[i] = o;
}

// ---------------- GEMM: C[M,ND] = A[M,ND] * W[ND,ND]^T (bf16, f32 acc) ---
// Wave tile: 64(M) x 64(N), 16 accumulators; 16 WMMAs per 16 b128 loads.
// mode 0: out bf16 [M,ND]      (Q, K, attention-O)
// mode 1: out bf16 V^T per batch: Vt[b][col][s]  (for PV B-operand loads)
// mode 2: out f32  [M,ND]      (final projection -> d_out)
__global__ __launch_bounds__(128) void gemm_xwT(const __bf16* __restrict__ A,
                                                const __bf16* __restrict__ W,
                                                void* __restrict__ outp,
                                                int mode) {
  const int lane = threadIdx.x & 31;
  const int wid  = threadIdx.x >> 5;      // 4 waves / block
  const int hl   = lane >> 4;             // lane-half
  const int l16  = lane & 15;
  const int nbase = (blockIdx.x * 4 + wid) * 64;   // N tile of 64
  const int m0    = blockIdx.y * 64;               // M tile of 64 (4 sub-tiles)
  const __bf16* Ap = A + (size_t)(m0 + l16) * ND + hl * 8;
  const __bf16* Wp = W + (size_t)(nbase + l16) * ND + hl * 8;

  v8f acc[4][4];
#pragma unroll
  for (int mi = 0; mi < 4; ++mi)
#pragma unroll
    for (int ni = 0; ni < 4; ++ni) acc[mi][ni] = (v8f){};

  for (int k = 0; k < ND; k += 32) {
    v16bf a[4], b[4];
#pragma unroll
    for (int mi = 0; mi < 4; ++mi) {
      const __bf16* p = Ap + (size_t)mi * 16 * ND + k;
      v8bf lo = *(const v8bf*)p;
      v8bf hi = *(const v8bf*)(p + 16);
#pragma unroll
      for (int i = 0; i < 8; ++i) { a[mi][i] = lo[i]; a[mi][i + 8] = hi[i]; }
    }
#pragma unroll
    for (int ni = 0; ni < 4; ++ni)
    {
      v8bf blo = *(const v8bf*)(Wp + (size_t)ni * 16 * ND + k);
      v8bf bhi = *(const v8bf*)(Wp + (size_t)ni * 16 * ND + k + 16);
#pragma unroll
      for (int i = 0; i < 8; ++i) { b[ni][i] = blo[i]; b[ni][i + 8] = bhi[i]; }
    }
    if (k + 32 < ND) {
      __builtin_prefetch(Ap + k + 32, 0, 1);
      __builtin_prefetch(Wp + k + 32, 0, 1);
    }
#pragma unroll
    for (int mi = 0; mi < 4; ++mi)
#pragma unroll
      for (int ni = 0; ni < 4; ++ni)
        acc[mi][ni] = wmma_bf16(a[mi], b[ni], acc[mi][ni]);
  }

  if (mode == 2) {
    float* Outf = (float*)outp;
#pragma unroll
    for (int mi = 0; mi < 4; ++mi)
#pragma unroll
      for (int ni = 0; ni < 4; ++ni) {
        int col = nbase + ni * 16 + l16;
#pragma unroll
        for (int r = 0; r < 8; ++r)
          Outf[(size_t)(m0 + mi * 16 + hl * 8 + r) * ND + col] = acc[mi][ni][r];
      }
  } else if (mode == 1) {
    __bf16* Vt = (__bf16*)outp;
#pragma unroll
    for (int mi = 0; mi < 4; ++mi) {
      int row0 = m0 + mi * 16 + hl * 8;     // 8 consecutive rows, same batch
      int b = row0 >> 11, s0 = row0 & (NS - 1);
#pragma unroll
      for (int ni = 0; ni < 4; ++ni) {
        int col = nbase + ni * 16 + l16;
        v8bf ov;
#pragma unroll
        for (int r = 0; r < 8; ++r) ov[r] = (__bf16)acc[mi][ni][r];
        *(v8bf*)(Vt + (size_t)b * ND * NS + (size_t)col * NS + s0) = ov;
      }
    }
  } else {
    __bf16* Outb = (__bf16*)outp;
#pragma unroll
    for (int mi = 0; mi < 4; ++mi)
#pragma unroll
      for (int ni = 0; ni < 4; ++ni) {
        int col = nbase + ni * 16 + l16;
#pragma unroll
        for (int r = 0; r < 8; ++r)
          Outb[(size_t)(m0 + mi * 16 + hl * 8 + r) * ND + col] = (__bf16)acc[mi][ni][r];
      }
  }
}

// ---------------- RoPE in-place on bf16 [NM, ND] ----------------
__global__ void rope_inplace(__bf16* __restrict__ buf, const int* __restrict__ pos) {
  int idx = blockIdx.x * blockDim.x + threadIdx.x;   // one rotation pair
  if (idx >= NM * (ND / 2)) return;
  int row = idx >> 9;          // / (ND/2)
  int pr  = idx & 511;
  int col = pr << 1;
  int dk  = col & (NDK - 1);   // position within the 64-wide head
  int s   = row & (NS - 1);
  float p   = (float)pos[s];
  float inv = __powf(10000.0f, -(float)dk * (1.0f / (float)NDK));
  float sn, cs;
  __sincosf(p * inv, &sn, &cs);
  size_t o = (size_t)row * ND + col;
  float x0 = (float)buf[o], x1 = (float)buf[o + 1];
  buf[o]     = (__bf16)(cs * x0 - sn * x1);
  buf[o + 1] = (__bf16)(sn * x0 + cs * x1);
}

// ---------------- Flash attention: 1 wave = 32 query rows ----------------
// K/V fragments loaded once per 32-key block, reused by both 16-row Q tiles:
// 16 WMMAs per 16 K/V b128 loads.
__global__ __launch_bounds__(128) void flash_fwd(const __bf16* __restrict__ Q,
                                                 const __bf16* __restrict__ K,
                                                 const __bf16* __restrict__ Vt,
                                                 __bf16* __restrict__ O) {
  __shared__ __bf16 Plds[4][2][16][32];  // per-wave, per-qtile P staging
  const int lane = threadIdx.x & 31;
  const int wid  = threadIdx.x >> 5;
  const int hl   = lane >> 4, l16 = lane & 15;
  const int q0   = (blockIdx.x * 4 + wid) * 32;
  const int h = blockIdx.y, b = blockIdx.z;
  const int hd0 = h * NDK;

  // Q A-fragments: 2 q-tiles x 2 K32 chunks
  v16bf qa[2][2];
#pragma unroll
  for (int t = 0; t < 2; ++t) {
    const size_t qoff = ((size_t)b * NS + q0 + t * 16 + l16) * ND + hd0;
#pragma unroll
    for (int kc = 0; kc < 2; ++kc) {
      v8bf lo = *(const v8bf*)(Q + qoff + kc * 32 + hl * 8);
      v8bf hi = *(const v8bf*)(Q + qoff + kc * 32 + 16 + hl * 8);
#pragma unroll
      for (int i = 0; i < 8; ++i) { qa[t][kc][i] = lo[i]; qa[t][kc][i + 8] = hi[i]; }
    }
  }

  v8f oacc[2][4];
#pragma unroll
  for (int t = 0; t < 2; ++t)
#pragma unroll
    for (int c = 0; c < 4; ++c) oacc[t][c] = (v8f){};
  float mrow[2][8], lrow[2][8];
#pragma unroll
  for (int t = 0; t < 2; ++t)
#pragma unroll
    for (int r = 0; r < 8; ++r) { mrow[t][r] = -1e30f; lrow[t][r] = 0.0f; }
  const float sc = 0.125f * 1.44269504088896340736f;  // 1/sqrt(64) * log2(e)
  const __bf16* Kbp = K + (size_t)b * NS * ND + hd0;
  const __bf16* Vbp = Vt + (size_t)b * ND * NS;

  for (int kv = 0; kv <= q0; kv += 32) {
    // ---- K fragments (shared by both q-tiles) ----
    v16bf kb[2][2];
#pragma unroll
    for (int jj = 0; jj < 2; ++jj) {
      const __bf16* kp = Kbp + (size_t)(kv + jj * 16 + l16) * ND + hl * 8;
      v8bf k0 = *(const v8bf*)(kp), k1 = *(const v8bf*)(kp + 16);
      v8bf k2 = *(const v8bf*)(kp + 32), k3 = *(const v8bf*)(kp + 48);
#pragma unroll
      for (int i = 0; i < 8; ++i) {
        kb[jj][0][i] = k0[i]; kb[jj][0][i + 8] = k1[i];
        kb[jj][1][i] = k2[i]; kb[jj][1][i + 8] = k3[i];
      }
    }
    // ---- scores: 2 q-tiles x 2 k-tiles ----
    v8f sfr[2][2];
#pragma unroll
    for (int t = 0; t < 2; ++t)
#pragma unroll
      for (int jj = 0; jj < 2; ++jj) {
        sfr[t][jj] = (v8f){};
        if (kv + jj * 16 <= q0 + t * 16 + 15) {   // tile has any unmasked column
          sfr[t][jj] = wmma_bf16(qa[t][0], kb[jj][0], sfr[t][jj]);
          sfr[t][jj] = wmma_bf16(qa[t][1], kb[jj][1], sfr[t][jj]);
        }
      }
    // ---- causal mask + online softmax (rows live in 16-lane groups) ----
#pragma unroll
    for (int t = 0; t < 2; ++t) {
#pragma unroll
      for (int r = 0; r < 8; ++r) {
        int rowq = q0 + t * 16 + hl * 8 + r;
        float a0 = (kv + l16      <= rowq) ? sfr[t][0][r] * sc : -1e30f;
        float a1 = (kv + 16 + l16 <= rowq) ? sfr[t][1][r] * sc : -1e30f;
        float mx = fmaxf(a0, a1);
#pragma unroll
        for (int off = 1; off < 16; off <<= 1) mx = fmaxf(mx, __shfl_xor(mx, off, 32));
        float mn = fmaxf(mrow[t][r], mx);
        float al = exp2f(mrow[t][r] - mn);
        float p0 = exp2f(a0 - mn);
        float p1 = exp2f(a1 - mn);
        float rs = p0 + p1;
#pragma unroll
        for (int off = 1; off < 16; off <<= 1) rs += __shfl_xor(rs, off, 32);
        lrow[t][r] = lrow[t][r] * al + rs;
        mrow[t][r] = mn;
        oacc[t][0][r] *= al; oacc[t][1][r] *= al;
        oacc[t][2][r] *= al; oacc[t][3][r] *= al;
        Plds[wid][t][hl * 8 + r][l16]      = (__bf16)p0;  // C-layout -> row-major P
        Plds[wid][t][hl * 8 + r][16 + l16] = (__bf16)p1;
      }
    }
    // ---- V fragments (shared by both q-tiles) ----
    v16bf vb[4];
#pragma unroll
    for (int c = 0; c < 4; ++c)
      vb[c] = *(const v16bf*)(Vbp + (size_t)(hd0 + c * 16 + l16) * NS + kv + hl * 16);
    // ---- reload P as A-fragments, O += P * V ----
#pragma unroll
    for (int t = 0; t < 2; ++t) {
      v16bf pa = *(const v16bf*)&Plds[wid][t][l16][hl * 16];
#pragma unroll
      for (int c = 0; c < 4; ++c)
        oacc[t][c] = wmma_bf16(pa, vb[c], oacc[t][c]);
    }
  }
  // ---- normalize + store bf16 O [NM, ND] ----
#pragma unroll
  for (int t = 0; t < 2; ++t)
#pragma unroll
    for (int r = 0; r < 8; ++r) {
      float inv = lrow[t][r] > 0.0f ? 1.0f / lrow[t][r] : 0.0f;
      size_t ro = ((size_t)b * NS + q0 + t * 16 + hl * 8 + r) * ND + hd0 + l16;
      O[ro + 0]  = (__bf16)(oacc[t][0][r] * inv);
      O[ro + 16] = (__bf16)(oacc[t][1][r] * inv);
      O[ro + 32] = (__bf16)(oacc[t][2][r] * inv);
      O[ro + 48] = (__bf16)(oacc[t][3][r] * inv);
    }
}

extern "C" void kernel_launch(void* const* d_in, const int* in_sizes, int n_in,
                              void* d_out, int out_size, void* d_ws, size_t ws_size,
                              hipStream_t stream) {
  (void)in_sizes; (void)n_in; (void)out_size; (void)ws_size;
  const float* x   = (const float*)d_in[0];
  const int*   pos = (const int*)d_in[1];
  const float* wq  = (const float*)d_in[2];
  const float* wk  = (const float*)d_in[3];
  const float* wv  = (const float*)d_in[4];
  const float* wo  = (const float*)d_in[5];

  char* ws = (char*)d_ws;                       // 88 MiB used, fully rewritten
  __bf16* xb  = (__bf16*)(ws);                  // 16 MiB  x in bf16
  __bf16* wqb = (__bf16*)(ws + (16ull << 20));  //  2 MiB each
  __bf16* wkb = (__bf16*)(ws + (18ull << 20));
  __bf16* wvb = (__bf16*)(ws + (20ull << 20));
  __bf16* wob = (__bf16*)(ws + (22ull << 20));
  __bf16* Qb  = (__bf16*)(ws + (24ull << 20));  // 16 MiB
  __bf16* Kb  = (__bf16*)(ws + (40ull << 20));  // 16 MiB
  __bf16* Vt  = (__bf16*)(ws + (56ull << 20));  // 16 MiB (V transposed per batch)
  __bf16* Ob  = (__bf16*)(ws + (72ull << 20));  // 16 MiB

  // 1) converts
  {
    int n4 = NM * ND / 4;
    cvt_f32_to_bf16<<<(n4 + 255) / 256, 256, 0, stream>>>((const float4*)x, (ushort4*)xb, n4);
    int w4 = ND * ND / 4;
    cvt_f32_to_bf16<<<(w4 + 255) / 256, 256, 0, stream>>>((const float4*)wq, (ushort4*)wqb, w4);
    cvt_f32_to_bf16<<<(w4 + 255) / 256, 256, 0, stream>>>((const float4*)wk, (ushort4*)wkb, w4);
    cvt_f32_to_bf16<<<(w4 + 255) / 256, 256, 0, stream>>>((const float4*)wv, (ushort4*)wvb, w4);
    cvt_f32_to_bf16<<<(w4 + 255) / 256, 256, 0, stream>>>((const float4*)wo, (ushort4*)wob, w4);
  }
  // 2) QKV projections (V written transposed)
  dim3 gg(4, NM / 64);   // 4 blocks * 4 waves = 16 N-tiles; 128 M-tiles of 64
  gemm_xwT<<<gg, 128, 0, stream>>>(xb, wqb, (void*)Qb, 0);
  gemm_xwT<<<gg, 128, 0, stream>>>(xb, wkb, (void*)Kb, 0);
  gemm_xwT<<<gg, 128, 0, stream>>>(xb, wvb, (void*)Vt, 1);
  // 3) RoPE on Q and K
  int np = NM * (ND / 2);
  rope_inplace<<<(np + 255) / 256, 256, 0, stream>>>(Qb, pos);
  rope_inplace<<<(np + 255) / 256, 256, 0, stream>>>(Kb, pos);
  // 4) causal flash attention (32 q-rows per wave)
  flash_fwd<<<dim3(NS / 128, NH, NB), 128, 0, stream>>>(Qb, Kb, Vt, Ob);
  // 5) output projection -> f32 d_out
  gemm_xwT<<<gg, 128, 0, stream>>>(Ob, wob, d_out, 2);
}
